// MultiHeadAttention_59794534695358
// MI455X (gfx1250) — compile-verified, matrix operand order checked by reading, not run on hardware
//
#include <hip/hip_runtime.h>
#include <hip/hip_bf16.h>
#include <stdint.h>

#define S_LEN 4096
#define DM    768
#define NH    12
#define HD    64
#define NB    2
#define MROWS (NB * S_LEN)   // 8192

typedef _Float16 v16h __attribute__((ext_vector_type(16)));
typedef _Float16 v8h  __attribute__((ext_vector_type(8)));
typedef float    v8f  __attribute__((ext_vector_type(8)));
typedef int      v4i  __attribute__((ext_vector_type(4)));

typedef __attribute__((address_space(1))) v4i* gv4i_p;
typedef __attribute__((address_space(3))) v4i* lv4i_p;

#if __has_builtin(__builtin_amdgcn_global_load_async_to_lds_b128)
#define HAVE_ASYNC_LDS 1
#else
#define HAVE_ASYNC_LDS 0
#endif

static __device__ __forceinline__ v16h cat16(v8h lo, v8h hi) {
    return __builtin_shufflevector(lo, hi, 0,1,2,3,4,5,6,7,8,9,10,11,12,13,14,15);
}
static __device__ __forceinline__ v8f wmma16(v16h a, v16h b, v8f c) {
    // D = A(16x32 f16) * B(32x16 f16) + C(16x16 f32)
    return __builtin_amdgcn_wmma_f32_16x16x32_f16(false, a, false, b, (short)0, c, false, false);
}

// 16-byte global -> LDS copy. Prefers the CDNA5 async data mover path
// (ASYNCcnt-tracked, bypasses VGPRs); falls back to a VGPR round trip.
static __device__ __forceinline__ void cp16_g2l(const _Float16* __restrict__ g,
                                                _Float16* l) {
#if HAVE_ASYNC_LDS
    __builtin_amdgcn_global_load_async_to_lds_b128(
        (gv4i_p)(void*)const_cast<_Float16*>(g),
        (lv4i_p)(void*)l, 0, 0);
#else
    *(v8h*)l = *(const v8h*)g;
#endif
}
static __device__ __forceinline__ void wait_async() {
#if HAVE_ASYNC_LDS
#if __has_builtin(__builtin_amdgcn_s_wait_asynccnt)
    __builtin_amdgcn_s_wait_asynccnt(0);
#else
    asm volatile("s_wait_asynccnt 0" ::: "memory");
#endif
#endif
}

// ---------------------------------------------------------------------------
// f32 -> f16 conversion kernels
// ---------------------------------------------------------------------------
__global__ __launch_bounds__(256) void cvt_f32_f16(const float* __restrict__ in,
                                                   _Float16* __restrict__ out, int n) {
    int i = blockIdx.x * 256 + threadIdx.x;
    if (i < n) out[i] = (_Float16)in[i];
}

// W stored [k(DM), n(DM)] row-major; write Wt[n][k] (f16) so WMMA B-fragments
// read contiguous K.
__global__ __launch_bounds__(256) void cvt_wt_f16(const float* __restrict__ W,
                                                  _Float16* __restrict__ Wt) {
    int i = blockIdx.x * 256 + threadIdx.x;
    if (i < DM * DM) {
        int k = i / DM, n = i % DM;
        Wt[(size_t)n * DM + k] = (_Float16)W[i];
    }
}

// ---------------------------------------------------------------------------
// Tiled WMMA GEMM: C[M=8192, N=768] = A(f16, row-major, K=768) @ Bt^T + bias
// Bt is [n][k] row-major f16. Block tile 128x128, 8 waves, each 32x64.
// OUTMODE 0: f16, write [b,h,s,d] head layout (Q, K)
// OUTMODE 1: f16, write [b,h,d,s] transposed head layout (V)
// OUTMODE 2: f32, write row-major [M, DM] (final output)
// ---------------------------------------------------------------------------
template <int OUTMODE>
__global__ __launch_bounds__(256) void gemm_wmma(const _Float16* __restrict__ A,
                                                 const _Float16* __restrict__ Bt,
                                                 const float* __restrict__ bias,
                                                 void* __restrict__ outp) {
    __shared__ __align__(16) _Float16 As[128 * 32];
    __shared__ __align__(16) _Float16 Bs[128 * 32];

    const int tid  = threadIdx.x;
    const int lane = tid & 31;
    const int w    = tid >> 5;   // 0..7
    const int wm   = w & 3;      // M quadrant (32 rows)
    const int wn   = w >> 2;     // N half (64 cols)
    const int l16  = lane & 15;
    const int hlf  = lane >> 4;
    const int m0   = blockIdx.x * 128;
    const int n0   = blockIdx.y * 128;

    const v8f vzero = {0.f, 0.f, 0.f, 0.f, 0.f, 0.f, 0.f, 0.f};
    v8f acc[2][4];
#pragma unroll
    for (int i = 0; i < 2; i++)
#pragma unroll
        for (int j = 0; j < 4; j++) acc[i][j] = vzero;

    for (int k0 = 0; k0 < DM; k0 += 32) {
        // Stage A-tile (128x32) and B-tile (128n x 32k): 512 chunks of 8 f16.
        for (int c = tid; c < 512; c += 256) {
            int row = c >> 2, col = (c & 3) << 3;
            cp16_g2l(&A [(size_t)(m0 + row) * DM + k0 + col], &As[row * 32 + col]);
            cp16_g2l(&Bt[(size_t)(n0 + row) * DM + k0 + col], &Bs[row * 32 + col]);
        }
        wait_async();
        __syncthreads();

        v16h af[2];
#pragma unroll
        for (int mt = 0; mt < 2; mt++) {
            const _Float16* p = &As[(wm * 32 + mt * 16 + l16) * 32 + hlf * 8];
            af[mt] = cat16(*(const v8h*)p, *(const v8h*)(p + 16));
        }
#pragma unroll
        for (int nt = 0; nt < 4; nt++) {
            const _Float16* p = &Bs[(wn * 64 + nt * 16 + l16) * 32 + hlf * 8];
            v16h bf = cat16(*(const v8h*)p, *(const v8h*)(p + 16));
#pragma unroll
            for (int mt = 0; mt < 2; mt++)
                acc[mt][nt] = wmma16(af[mt], bf, acc[mt][nt]);
        }
        __syncthreads();
    }

    // Epilogue: bias add + store.
#pragma unroll
    for (int nt = 0; nt < 4; nt++) {
        int   gn = n0 + wn * 64 + nt * 16 + l16;
        float bv = bias[gn];
#pragma unroll
        for (int mt = 0; mt < 2; mt++) {
#pragma unroll
            for (int r = 0; r < 8; r++) {
                int   gm = m0 + wm * 32 + mt * 16 + hlf * 8 + r;
                float v  = acc[mt][nt][r] + bv;
                if constexpr (OUTMODE == 2) {
                    ((float*)outp)[(size_t)gm * DM + gn] = v;
                } else {
                    int b = gm >> 12, s = gm & (S_LEN - 1);
                    int h = gn >> 6,  d = gn & 63;
                    _Float16* o = (_Float16*)outp;
                    if constexpr (OUTMODE == 0)
                        o[((size_t)(b * NH + h) * S_LEN + s) * HD + d] = (_Float16)v;
                    else
                        o[((size_t)(b * NH + h) * HD + d) * S_LEN + s] = (_Float16)v;
                }
            }
        }
    }
}

// ---------------------------------------------------------------------------
// Flash attention. Block = one (b,h) x 128 query rows; 8 waves x 16 rows.
// Q [bh][s][d] f16, K [bh][s][d] f16, Vt [bh][d][s] f16, AO f16 [8192][768].
// Row sums of P are accumulated with WMMA (P @ ones) -> same C-layout as O,
// so the final normalization needs no cross-lane reduction.
// ---------------------------------------------------------------------------
__global__ __launch_bounds__(256) void flash_attn(const _Float16* __restrict__ Q,
                                                  const _Float16* __restrict__ K,
                                                  const _Float16* __restrict__ Vt,
                                                  _Float16* __restrict__ AO) {
    __shared__ __align__(16) _Float16 Ks[64 * 64];       // [key][d]
    __shared__ __align__(16) _Float16 Vs[64 * 64];       // [d][key]
    __shared__ __align__(16) _Float16 Ps[8 * 16 * 64];   // per-wave P scratch

    const int tid  = threadIdx.x;
    const int lane = tid & 31;
    const int w    = tid >> 5;
    const int l16  = lane & 15;
    const int hlf  = lane >> 4;
    const int bh   = blockIdx.y;        // 0..23
    const int q0   = blockIdx.x * 128;  // query tile base

    const _Float16* Qb = Q  + (size_t)bh * S_LEN * HD;
    const _Float16* Kb = K  + (size_t)bh * S_LEN * HD;
    const _Float16* Vb = Vt + (size_t)bh * HD * S_LEN;

    // Resident Q fragments (16 rows x 64 d), pre-scaled by 1/sqrt(HD)=0.125
    // (exact power of two in f16).
    v16h qf[2];
    {
        int qr = q0 + w * 16 + l16;
#pragma unroll
        for (int kk = 0; kk < 2; kk++) {
            const _Float16* p = Qb + (size_t)qr * HD + kk * 32 + hlf * 8;
            v16h t = cat16(*(const v8h*)p, *(const v8h*)(p + 16));
#pragma unroll
            for (int e = 0; e < 16; e++) t[e] = t[e] * (_Float16)0.125f;
            qf[kk] = t;
        }
    }

    const v8f vzero = {0.f, 0.f, 0.f, 0.f, 0.f, 0.f, 0.f, 0.f};
    v16h onesf;
#pragma unroll
    for (int e = 0; e < 16; e++) onesf[e] = (_Float16)1.0f;

    v8f o[4];
#pragma unroll
    for (int nt = 0; nt < 4; nt++) o[nt] = vzero;
    v8f   lacc = vzero;   // per-row softmax denominators (C-layout, replicated)
    float mrow[8];
#pragma unroll
    for (int r = 0; r < 8; r++) mrow[r] = -1e30f;

    for (int s0 = 0; s0 < S_LEN; s0 += 64) {
        // Stage 64-key chunk of K (row-major key x d) and Vt (d x key).
        for (int c = tid; c < 512; c += 256) {
            int row = c >> 3, col = (c & 7) << 3;
            cp16_g2l(&Kb[(size_t)(s0 + row) * HD + col], &Ks[row * 64 + col]);
            cp16_g2l(&Vb[(size_t)row * S_LEN + s0 + col], &Vs[row * 64 + col]);
        }
        wait_async();
        __syncthreads();

        // S = (Q/sqrt(d)) @ K^T  (16 rows x 64 keys), 4 key-tiles x 2 K-steps.
        v8f sf[4];
#pragma unroll
        for (int nt = 0; nt < 4; nt++) sf[nt] = vzero;
#pragma unroll
        for (int nt = 0; nt < 4; nt++) {
#pragma unroll
            for (int kk = 0; kk < 2; kk++) {
                const _Float16* p = &Ks[(nt * 16 + l16) * 64 + kk * 32 + hlf * 8];
                v16h bf = cat16(*(const v8h*)p, *(const v8h*)(p + 16));
                sf[nt] = wmma16(qf[kk], bf, sf[nt]);
            }
        }

        // Online softmax: cross-lane max (16-lane halves), exp, rescale O and l.
#pragma unroll
        for (int r = 0; r < 8; r++) {
            float mx = fmaxf(fmaxf(sf[0][r], sf[1][r]), fmaxf(sf[2][r], sf[3][r]));
#pragma unroll
            for (int msk = 1; msk < 16; msk <<= 1)
                mx = fmaxf(mx, __shfl_xor(mx, msk, 32));
            float mn    = fmaxf(mrow[r], mx);
            float alpha = __expf(mrow[r] - mn);
            mrow[r] = mn;
            lacc[r] *= alpha;
#pragma unroll
            for (int nt = 0; nt < 4; nt++) o[nt][r] *= alpha;
#pragma unroll
            for (int nt = 0; nt < 4; nt++)
                sf[nt][r] = __expf(sf[nt][r] - mn);
        }

        // C-layout -> A-layout transpose of P through per-wave LDS scratch.
        _Float16* Pw = &Ps[w * 16 * 64];
#pragma unroll
        for (int nt = 0; nt < 4; nt++)
#pragma unroll
            for (int r = 0; r < 8; r++)
                Pw[(hlf * 8 + r) * 64 + nt * 16 + l16] = (_Float16)sf[nt][r];

        asm volatile("s_wait_dscnt 0" ::: "memory");  // same-wave LDS store->load ordering

        // O += P @ V ; l += P @ ones (row sums, reusing the P fragments).
#pragma unroll
        for (int kk = 0; kk < 2; kk++) {
            const _Float16* pp = &Pw[l16 * 64 + kk * 32 + hlf * 8];
            v16h pf = cat16(*(const v8h*)pp, *(const v8h*)(pp + 16));
            lacc = wmma16(pf, onesf, lacc);
#pragma unroll
            for (int nt = 0; nt < 4; nt++) {
                const _Float16* vp = &Vs[(nt * 16 + l16) * 64 + kk * 32 + hlf * 8];
                v16h bf = cat16(*(const v8h*)vp, *(const v8h*)(vp + 16));
                o[nt] = wmma16(pf, bf, o[nt]);
            }
        }
        __syncthreads();
    }

    // Epilogue: normalize by per-row denominator, store f16 into [B*S, D].
    const int b = bh / NH, h = bh % NH;
#pragma unroll
    for (int nt = 0; nt < 4; nt++) {
#pragma unroll
        for (int r = 0; r < 8; r++) {
            int srow = q0 + w * 16 + hlf * 8 + r;
            int gm   = b * S_LEN + srow;
            int gn   = h * HD + nt * 16 + l16;
            AO[(size_t)gm * DM + gn] = (_Float16)(o[nt][r] / lacc[r]);
        }
    }
}

// ---------------------------------------------------------------------------
// Host-side orchestration
// ---------------------------------------------------------------------------
extern "C" void kernel_launch(void* const* d_in, const int* in_sizes, int n_in,
                              void* d_out, int out_size, void* d_ws, size_t ws_size,
                              hipStream_t stream) {
    const float* x  = (const float*)d_in[0];
    const float* Wq = (const float*)d_in[1];
    const float* bq = (const float*)d_in[2];
    const float* Wk = (const float*)d_in[3];
    const float* bk = (const float*)d_in[4];
    const float* Wv = (const float*)d_in[5];
    const float* bv = (const float*)d_in[6];
    const float* Wo = (const float*)d_in[7];
    const float* bo = (const float*)d_in[8];
    float* out = (float*)d_out;

    char*  ws  = (char*)d_ws;
    size_t off = 0;
    _Float16* Xh  = (_Float16*)(ws + off); off += (size_t)MROWS * DM * 2;
    _Float16* Wqt = (_Float16*)(ws + off); off += (size_t)DM * DM * 2;
    _Float16* Wkt = (_Float16*)(ws + off); off += (size_t)DM * DM * 2;
    _Float16* Wvt = (_Float16*)(ws + off); off += (size_t)DM * DM * 2;
    _Float16* Wot = (_Float16*)(ws + off); off += (size_t)DM * DM * 2;
    _Float16* Qh  = (_Float16*)(ws + off); off += (size_t)MROWS * DM * 2;
    _Float16* Kh  = (_Float16*)(ws + off); off += (size_t)MROWS * DM * 2;
    _Float16* Vth = (_Float16*)(ws + off); off += (size_t)MROWS * DM * 2;
    _Float16* AOh = (_Float16*)(ws + off); off += (size_t)MROWS * DM * 2;

    const int nX = MROWS * DM;
    cvt_f32_f16<<<(nX + 255) / 256, 256, 0, stream>>>(x, Xh, nX);
    const int wblocks = (DM * DM + 255) / 256;
    cvt_wt_f16<<<wblocks, 256, 0, stream>>>(Wq, Wqt);
    cvt_wt_f16<<<wblocks, 256, 0, stream>>>(Wk, Wkt);
    cvt_wt_f16<<<wblocks, 256, 0, stream>>>(Wv, Wvt);
    cvt_wt_f16<<<wblocks, 256, 0, stream>>>(Wo, Wot);

    dim3 gg(MROWS / 128, DM / 128);  // 64 x 6
    gemm_wmma<0><<<gg, 256, 0, stream>>>(Xh, Wqt, bq, (void*)Qh);
    gemm_wmma<0><<<gg, 256, 0, stream>>>(Xh, Wkt, bk, (void*)Kh);
    gemm_wmma<1><<<gg, 256, 0, stream>>>(Xh, Wvt, bv, (void*)Vth);

    flash_attn<<<dim3(S_LEN / 128, NB * NH), 256, 0, stream>>>(Qh, Kh, Vth, AOh);

    gemm_wmma<2><<<gg, 256, 0, stream>>>(AOh, Wot, bo, (void*)out);
}
